// DynamicModel_78340203479583
// MI455X (gfx1250) — matrix operand order repaired and checked by reading, not compiled, not run on hardware
//
#include <hip/hip_runtime.h>

// ---------------------------------------------------------------------------
// Problem constants (hardcoded in the reference module)
// ---------------------------------------------------------------------------
static constexpr int P = 50, Q = 10, M = 100;

// Padded GEMM dims (N -> multiple of 64, K -> multiple of 32)
static constexpr int K1  = P * M,  K1p = 5024, N1 = P * M, N1p = 5056;  // w: e1 @ Ww^T
static constexpr int K2  = Q * M,  K2p = 1024, N2 = Q * M, N2p = 1024;  // u/w2/u2
static constexpr int Kv  = P * Q,  Kvp = 512,  Nv = P * Q, Nvp = 512;   // v
static constexpr int Kv2 = Q * Q,  Kv2p = 128, Nv2 = Q * Q, Nv2p = 128; // v2
static constexpr int Kf1 = 3 * M,  Kf1p = 320, Nf1 = 1000, Nf1p = 1024; // fc1
static constexpr int Kf2 = 1000,   Kf2p = 1024, Nf2 = 800, Nf2p = 832;  // fc2

typedef __attribute__((ext_vector_type(16))) _Float16 v16h;
typedef __attribute__((ext_vector_type(8)))  _Float16 v8h;
typedef __attribute__((ext_vector_type(8)))  float    v8f;

#define DEV_INLINE __device__ __forceinline__

// CDNA5 async global->LDS copy path.
// Signature (from compiler diagnostic): param0 = int4 in AS1 (global),
// param1 = int4 in AS3 (LDS), then imm offset, imm cpol.
#if defined(__gfx1250__) && __has_builtin(__builtin_amdgcn_global_load_async_to_lds_b128) && \
    __has_builtin(__builtin_amdgcn_s_wait_asynccnt)
#define USE_ASYNC_LDS 1
typedef int v4i_vs __attribute__((vector_size(16)));
typedef __attribute__((address_space(1))) v4i_vs GV4;
typedef __attribute__((address_space(3))) v4i_vs LV4;
// copy one 16B chunk global -> LDS (per active lane)
#define ASYNC_COPY_CHUNK(dstLds, srcGlb) \
  __builtin_amdgcn_global_load_async_to_lds_b128((GV4*)(srcGlb), (LV4*)(dstLds), 0, 0)
#define ASYNC_WAIT() __builtin_amdgcn_s_wait_asynccnt(0)
#else
#define USE_ASYNC_LDS 0
#endif

// ---------------------------------------------------------------------------
// WMMA fragment helpers (wave32; A and B use the same contiguous k order)
// ---------------------------------------------------------------------------
// A fragment: 16x32 f16 tile, row-major source (rows m, cols k), stride lda.
// lanes 0-15: row=lane, K = 0..15; lanes 16-31: row=lane-16, K = 16..31
DEV_INLINE v16h load_frag_a(const _Float16* __restrict__ a, int lda) {
  int lane = threadIdx.x & 31;
  const _Float16* p = a + (size_t)(lane & 15) * lda + ((lane >> 4) ? 16 : 0);
  v8h lo = *(const v8h*)p;          // K = khalf .. khalf+7
  v8h hi = *(const v8h*)(p + 8);    // K = khalf+8 .. khalf+15
  v16h f;
#pragma unroll
  for (int i = 0; i < 8; ++i) { f[i] = lo[i]; f[8 + i] = hi[i]; }
  return f;
}

// B fragment for X @ W^T: W is row-major [N,K]; B[k][n] = W[n][k].
// lanes 0-15: col=lane, K = 0..15; lanes 16-31: col=lane-16, K = 16..31
DEV_INLINE v16h load_frag_b(const _Float16* __restrict__ w, int ldw) {
  int lane = threadIdx.x & 31;
  const _Float16* p = w + (size_t)(lane & 15) * ldw + ((lane >> 4) ? 16 : 0);
  return *(const v16h*)p;           // contiguous 32B, alignment guaranteed
}

// ---------------------------------------------------------------------------
// Generic batched GEMM: Y[g][m,n] = X[g][m,:] . W[g][n,:] + bias[g][n]
// Block = 128 threads (4 waves); block tile 128(M) x 64(N); wave = 32x64 stripe
// (2 A-fragments x 4 B-fragments -> 8 WMMAs per K-step, B-frag reused twice).
// ---------------------------------------------------------------------------
template <bool RELU, bool OUT_F16>
__global__ __launch_bounds__(128) void k_gemm_f16(
    const _Float16* __restrict__ A, long long ag, int lda,
    const _Float16* __restrict__ W, long long wg, int ldw,
    const float* __restrict__ bias, long long bg, int nbias,
    void* __restrict__ Y, long long yg, int ldy, int nstore, int Kp) {
  int g = blockIdx.z;
  A += (size_t)g * ag;
  W += (size_t)g * wg;
  const float* bptr = bias ? bias + (size_t)g * bg : nullptr;

  int wave = threadIdx.x >> 5;
  int lane = threadIdx.x & 31;
  int m0 = blockIdx.y * 128 + wave * 32;     // rows m0..m0+31
  int n0 = blockIdx.x * 64;
  int col = lane & 15;
  int mbase = (lane >> 4) * 8;

  v8f acc0[4], acc1[4];
#pragma unroll
  for (int t = 0; t < 4; ++t) {
    int c = n0 + t * 16 + col;
    float bv = (bptr && c < nbias) ? bptr[c] : 0.f;
#pragma unroll
    for (int r = 0; r < 8; ++r) { acc0[t][r] = bv; acc1[t][r] = bv; }
  }

  const _Float16* Ap0 = A + (size_t)m0 * lda;
  const _Float16* Ap1 = Ap0 + (size_t)16 * lda;
  for (int k = 0; k < Kp; k += 32) {
    v16h a0 = load_frag_a(Ap0 + k, lda);
    v16h a1 = load_frag_a(Ap1 + k, lda);
#pragma unroll
    for (int t = 0; t < 4; ++t) {
      v16h bf = load_frag_b(W + (size_t)(n0 + t * 16) * ldw + k, ldw);
      acc0[t] = __builtin_amdgcn_wmma_f32_16x16x32_f16(
          false, a0, false, bf, (short)0, acc0[t], false, false);
      acc1[t] = __builtin_amdgcn_wmma_f32_16x16x32_f16(
          false, a1, false, bf, (short)0, acc1[t], false, false);
    }
  }

#pragma unroll
  for (int t = 0; t < 4; ++t) {
    int c = n0 + t * 16 + col;
    if (c < nstore) {
#pragma unroll
      for (int r = 0; r < 8; ++r) {
        float v0 = acc0[t][r], v1 = acc1[t][r];
        if (RELU) { v0 = v0 > 0.f ? v0 : 0.f; v1 = v1 > 0.f ? v1 : 0.f; }
        size_t base = (size_t)((size_t)g * (size_t)yg) + c;
        size_t o0 = base + (size_t)(m0 + mbase + r) * ldy;
        size_t o1 = base + (size_t)(m0 + 16 + mbase + r) * ldy;
        if (OUT_F16) {
          ((_Float16*)Y)[o0] = (_Float16)v0;
          ((_Float16*)Y)[o1] = (_Float16)v1;
        } else {
          ((float*)Y)[o0] = v0;
          ((float*)Y)[o1] = v1;
        }
      }
    }
  }
}

// ---------------------------------------------------------------------------
// f32 [G][N,K] -> zero-padded f16 [G][Np,Kp]
// ---------------------------------------------------------------------------
__global__ void k_convert_pad_f16(const float* __restrict__ W, _Float16* __restrict__ Wh,
                                  int N, int K, int Np, int Kp) {
  int g = blockIdx.z;
  int j = blockIdx.x * blockDim.x + threadIdx.x;
  int i = blockIdx.y;
  if (j >= Kp) return;
  float v = (i < N && j < K) ? W[(size_t)g * N * K + (size_t)i * K + j] : 0.f;
  Wh[(size_t)g * Np * Kp + (size_t)i * Kp + j] = (_Float16)v;
}

// Embedding gather -> padded f16 e1h[b, k]  (k = p*M + m)
__global__ void k_gather_e1h(const int* __restrict__ x1, const float* __restrict__ emb,
                             _Float16* __restrict__ e1h, int Kp) {
  int b = blockIdx.y;
  int k = blockIdx.x * blockDim.x + threadIdx.x;
  if (k >= Kp) return;
  float v = 0.f;
  if (k < P * M) {
    int p = k / M, m = k - p * M;
    v = emb[(size_t)x1[b * P + p] * M + m];
  }
  e1h[(size_t)b * Kp + k] = (_Float16)v;
}

// ---------------------------------------------------------------------------
// Stage 1 co-attention: s[b,p,q] = mean_m tanh(w[b,p,m] + u[b,q,m]) -> f16 padded
// w/u rows staged in LDS via CDNA5 async global->LDS copies when available.
// ---------------------------------------------------------------------------
__global__ __launch_bounds__(512) void k_coattn1(const float* __restrict__ w_out, int ldw,
                                                 const float* __restrict__ u_out, int ldu,
                                                 _Float16* __restrict__ sh, int ldsY) {
  __shared__ __align__(16) float wbuf[P * M];   // 20 KB
  __shared__ __align__(16) float ubuf[Q * M];   //  4 KB
  int b = blockIdx.x;
  const float* wr = w_out + (size_t)b * ldw;
  const float* ur = u_out + (size_t)b * ldu;
#if USE_ASYNC_LDS
  // 16B chunks: 5000 floats = 1250 chunks; 1000 floats = 250 chunks (exact)
  for (int c = threadIdx.x; c < (P * M) / 4; c += blockDim.x)
    ASYNC_COPY_CHUNK(wbuf + c * 4, wr + c * 4);
  for (int c = threadIdx.x; c < (Q * M) / 4; c += blockDim.x)
    ASYNC_COPY_CHUNK(ubuf + c * 4, ur + c * 4);
  ASYNC_WAIT();
#else
  for (int i = threadIdx.x; i < P * M; i += blockDim.x) wbuf[i] = wr[i];
  for (int i = threadIdx.x; i < Q * M; i += blockDim.x) ubuf[i] = ur[i];
#endif
  __syncthreads();
  for (int idx = threadIdx.x; idx < ldsY; idx += blockDim.x) {
    float val = 0.f;
    if (idx < P * Q) {
      int p = idx / Q, q = idx - p * Q;
      float s = 0.f;
      for (int m = 0; m < M; ++m) s += tanhf(wbuf[p * M + m] + ubuf[q * M + m]);
      val = s * (1.f / M);
    }
    sh[(size_t)b * ldsY + idx] = (_Float16)val;
  }
}

// Masked softmax over q, then zero rows p >= x1_len
__global__ void k_alpha(const float* __restrict__ v_out, const int* __restrict__ x1len,
                        const int* __restrict__ x2len, float* __restrict__ alpha, int nBP) {
  int i = blockIdx.x * blockDim.x + threadIdx.x;
  if (i >= nBP) return;
  int b = i / P, p = i - b * P;
  int l2 = x2len[b];
  float rowmask = (p < x1len[b]) ? 1.f : 0.f;
  const float* vr = v_out + (size_t)b * Nvp + p * Q;
  float vals[Q], mx = -1e30f;
#pragma unroll
  for (int q = 0; q < Q; ++q) {
    float x = (q < l2) ? vr[q] : -1e9f;
    vals[q] = x; mx = fmaxf(mx, x);
  }
  float ssum = 0.f;
#pragma unroll
  for (int q = 0; q < Q; ++q) { float e = expf(vals[q] - mx); vals[q] = e; ssum += e; }
  float inv = rowmask / ssum;
#pragma unroll
  for (int q = 0; q < Q; ++q) alpha[(size_t)b * (P * Q) + p * Q + q] = vals[q] * inv;
}

// inp[b,q,m] = sum_p alpha[b,p,q] * e1[b,p,m]  (e1 gathered on the fly), mask q
__global__ __launch_bounds__(256) void k_inp(const int* __restrict__ x1,
                                             const float* __restrict__ emb,
                                             const float* __restrict__ alpha,
                                             const int* __restrict__ x2len,
                                             float* __restrict__ inp,
                                             _Float16* __restrict__ inph) {
  __shared__ __align__(16) float aL[P * Q];     // 2 KB
  __shared__ __align__(16) float eL[P * M];     // 20 KB
  __shared__ __align__(16) float sbuf[Q * M];   // 4 KB
  int b = blockIdx.x;
#if USE_ASYNC_LDS
  // alpha row: 500 floats = 125 chunks; e1 rows: 25 chunks of 16B per p (M=100 floats)
  const float* ar = alpha + (size_t)b * P * Q;
  for (int c = threadIdx.x; c < (P * Q) / 4; c += blockDim.x)
    ASYNC_COPY_CHUNK(aL + c * 4, ar + c * 4);
  for (int c = threadIdx.x; c < (P * M) / 4; c += blockDim.x) {
    int p = c / (M / 4), cc = c % (M / 4);
    const float* src = emb + (size_t)x1[b * P + p] * M + cc * 4;
    ASYNC_COPY_CHUNK(eL + p * M + cc * 4, src);
  }
  ASYNC_WAIT();
#else
  for (int i = threadIdx.x; i < P * Q; i += blockDim.x) aL[i] = alpha[(size_t)b * P * Q + i];
  for (int i = threadIdx.x; i < P * M; i += blockDim.x) {
    int p = i / M, m = i - p * M;
    eL[i] = emb[(size_t)x1[b * P + p] * M + m];
  }
#endif
  __syncthreads();
  int l2 = x2len[b];
  for (int idx = threadIdx.x; idx < Q * M; idx += blockDim.x) {
    int q = idx / M, m = idx - q * M;
    float s = 0.f;
#pragma unroll 10
    for (int p = 0; p < P; ++p) s += aL[p * Q + q] * eL[p * M + m];
    s *= (q < l2) ? 1.f : 0.f;
    sbuf[idx] = s;
    inp[(size_t)b * (Q * M) + idx] = s;
  }
  __syncthreads();
  for (int idx = threadIdx.x; idx < K2p; idx += blockDim.x)
    inph[(size_t)b * K2p + idx] = (_Float16)((idx < Q * M) ? sbuf[idx] : 0.f);
}

// s2[b,q] = mean_m tanh(w2 + u2)
__global__ void k_s2(const float* __restrict__ w2, const float* __restrict__ u2,
                     float* __restrict__ s2, int nBQ) {
  int i = blockIdx.x * blockDim.x + threadIdx.x;
  if (i >= nBQ) return;
  int b = i / Q, q = i - b * Q;
  const float* wr = w2 + (size_t)b * N2p + q * M;
  const float* ur = u2 + (size_t)b * N2p + q * M;
  float s = 0.f;
  for (int m = 0; m < M; ++m) s += tanhf(wr[m] + ur[m]);
  s2[i] = s * (1.f / M);
}

// s2mat flatten -> f16 padded: s2h[b, i*Q+j] = s2[b,i]
__global__ void k_s2h(const float* __restrict__ s2, _Float16* __restrict__ s2h) {
  int b = blockIdx.x, k = threadIdx.x;          // blockDim = Kv2p
  float v = 0.f;
  if (k < Q * Q) v = s2[(size_t)b * Q + k / Q];
  s2h[(size_t)b * Kv2p + k] = (_Float16)v;
}

// rowsum/softmax (alpha2), shared vec, x_mlp, and xg assembly (f16 padded)
__global__ __launch_bounds__(128) void k_stage2(const float* __restrict__ v2,
                                                const int* __restrict__ x2len,
                                                const float* __restrict__ inp,
                                                const float* __restrict__ x2,
                                                _Float16* __restrict__ xgh) {
  __shared__ float rs[Q];
  __shared__ float a2[Q];
  __shared__ float shv[M];
  int b = blockIdx.x;
  int l2 = x2len[b];
  if (threadIdx.x < Q) {
    int i = threadIdx.x;
    const float* vr = v2 + (size_t)b * Nv2p + i * Q;
    float s = 0.f;
#pragma unroll
    for (int j = 0; j < Q; ++j) s += (j < l2) ? vr[j] : 0.f;
    rs[i] = s - vr[i];                          // diagonal excluded unconditionally
  }
  __syncthreads();
  if (threadIdx.x == 0) {
    float mx = -1e30f;
#pragma unroll
    for (int i = 0; i < Q; ++i) { float x = (i < l2) ? rs[i] : -1e9f; rs[i] = x; mx = fmaxf(mx, x); }
    float ssum = 0.f;
#pragma unroll
    for (int i = 0; i < Q; ++i) { float e = expf(rs[i] - mx); a2[i] = e; ssum += e; }
    float inv = 1.f / ssum;
#pragma unroll
    for (int i = 0; i < Q; ++i) a2[i] *= inv;
  }
  __syncthreads();
  for (int m = threadIdx.x; m < M; m += blockDim.x) {
    float s = 0.f;
#pragma unroll
    for (int q = 0; q < Q; ++q) s += a2[q] * inp[(size_t)b * (Q * M) + q * M + m];
    shv[m] = s;
  }
  __syncthreads();
  for (int idx = threadIdx.x; idx < 10 * Kf1p; idx += blockDim.x) {
    int g = idx / Kf1p, c = idx - g * Kf1p;
    float v = 0.f;
    if (c < 3 * M) {
      int row = (g * 3 + c / M) % Q, m = c % M;
      size_t o = (size_t)b * (Q * M) + row * M + m;
      v = inp[o] + shv[m] + x2[o];
    }
    xgh[(size_t)b * (10 * Kf1p) + idx] = (_Float16)v;
  }
}

// in-place softmax over rows of length 800
__global__ __launch_bounds__(256) void k_softmax800(float* __restrict__ out) {
  __shared__ float sm[256];
  float* row = out + (size_t)blockIdx.x * Nf2;
  float mx = -1e30f;
  for (int i = threadIdx.x; i < Nf2; i += 256) mx = fmaxf(mx, row[i]);
  sm[threadIdx.x] = mx; __syncthreads();
  for (int s = 128; s > 0; s >>= 1) {
    if (threadIdx.x < s) sm[threadIdx.x] = fmaxf(sm[threadIdx.x], sm[threadIdx.x + s]);
    __syncthreads();
  }
  mx = sm[0]; __syncthreads();
  float ssum = 0.f;
  for (int i = threadIdx.x; i < Nf2; i += 256) { float e = expf(row[i] - mx); row[i] = e; ssum += e; }
  sm[threadIdx.x] = ssum; __syncthreads();
  for (int s = 128; s > 0; s >>= 1) {
    if (threadIdx.x < s) sm[threadIdx.x] += sm[threadIdx.x + s];
    __syncthreads();
  }
  float inv = 1.f / sm[0];
  for (int i = threadIdx.x; i < Nf2; i += 256) row[i] *= inv;
}

// ---------------------------------------------------------------------------
// Host orchestration
// ---------------------------------------------------------------------------
extern "C" void kernel_launch(void* const* d_in, const int* in_sizes, int n_in,
                              void* d_out, int out_size, void* d_ws, size_t ws_size,
                              hipStream_t stream) {
  (void)n_in; (void)out_size; (void)ws_size;
  const int*   x1    = (const int*)  d_in[0];
  const float* x2    = (const float*)d_in[1];
  const int*   x1len = (const int*)  d_in[2];
  const int*   x2len = (const int*)  d_in[3];
  const float* emb   = (const float*)d_in[5];
  const float* Ww    = (const float*)d_in[6];
  const float* bw    = (const float*)d_in[7];
  const float* Wu    = (const float*)d_in[8];
  const float* bu    = (const float*)d_in[9];
  const float* Wv    = (const float*)d_in[10];
  const float* bv    = (const float*)d_in[11];
  const float* Ww2   = (const float*)d_in[12];
  const float* bw2   = (const float*)d_in[13];
  const float* Wu2   = (const float*)d_in[14];
  const float* bu2   = (const float*)d_in[15];
  const float* Wv2   = (const float*)d_in[16];
  const float* bv2   = (const float*)d_in[17];
  const float* Wfc1  = (const float*)d_in[18];
  const float* bfc1  = (const float*)d_in[19];
  const float* Wfc2  = (const float*)d_in[20];
  const float* bfc2  = (const float*)d_in[21];
  float* out = (float*)d_out;

  const int B = in_sizes[0] / P;   // 2048 (multiple of 128)

  // --- workspace bump allocator (256B aligned) ---
  char*  ws  = (char*)d_ws;
  size_t off = 0;
  auto alloc = [&](size_t bytes) -> char* {
    off = (off + 255) & ~(size_t)255;
    char* p = ws + off;
    off += bytes;
    return p;
  };
  _Float16* Wwh   = (_Float16*)alloc((size_t)N1p * K1p * 2);
  _Float16* Wuh   = (_Float16*)alloc((size_t)N2p * K2p * 2);
  _Float16* Ww2h  = (_Float16*)alloc((size_t)N2p * K2p * 2);
  _Float16* Wu2h  = (_Float16*)alloc((size_t)N2p * K2p * 2);
  _Float16* Wvh   = (_Float16*)alloc((size_t)Nvp * Kvp * 2);
  _Float16* Wv2h  = (_Float16*)alloc((size_t)Nv2p * Kv2p * 2);
  _Float16* Wfc1h = (_Float16*)alloc((size_t)10 * Nf1p * Kf1p * 2);
  _Float16* Wfc2h = (_Float16*)alloc((size_t)10 * Nf2p * Kf2p * 2);
  _Float16* e1h   = (_Float16*)alloc((size_t)B * K1p * 2);
  _Float16* x2h   = (_Float16*)alloc((size_t)B * K2p * 2);
  float*    w_out = (float*)   alloc((size_t)B * N1p * 4);
  float*    u_out = (float*)   alloc((size_t)B * N2p * 4);
  _Float16* sh    = (_Float16*)alloc((size_t)B * Kvp * 2);
  float*    v_out = (float*)   alloc((size_t)B * Nvp * 4);
  float*    alpha = (float*)   alloc((size_t)B * P * Q * 4);
  float*    inp   = (float*)   alloc((size_t)B * Q * M * 4);
  _Float16* inph  = (_Float16*)alloc((size_t)B * K2p * 2);
  float*    w2o   = (float*)   alloc((size_t)B * N2p * 4);
  float*    u2o   = (float*)   alloc((size_t)B * N2p * 4);
  float*    s2    = (float*)   alloc((size_t)B * Q * 4);
  _Float16* s2h   = (_Float16*)alloc((size_t)B * Kv2p * 2);
  float*    v2o   = (float*)   alloc((size_t)B * Nv2p * 4);
  // Aliases onto dead buffers (lifetimes are disjoint):
  _Float16* xgh = (_Float16*)e1h;    // B*10*320 f16 (12.5MB) <= e1h (19.6MB); e1h dead after first GEMM
  _Float16* h1h = (_Float16*)w_out;  // B*10*1024 f16 (40MB) <= w_out+u_out (47.5MB); both dead by then

  dim3 b256(256), b128(128), b512(512);
  auto cvt = [&](const float* src, _Float16* dst, int N, int K, int Np, int Kp, int G) {
    dim3 g((Kp + 255) / 256, Np, G);
    k_convert_pad_f16<<<g, b256, 0, stream>>>(src, dst, N, K, Np, Kp);
  };

  // --- weight + activation conversion (per-launch; ~250MB HBM ~= 10us) ---
  cvt(Ww,   Wwh,   N1, K1, N1p, K1p, 1);
  cvt(Wu,   Wuh,   N2, K2, N2p, K2p, 1);
  cvt(Ww2,  Ww2h,  N2, K2, N2p, K2p, 1);
  cvt(Wu2,  Wu2h,  N2, K2, N2p, K2p, 1);
  cvt(Wv,   Wvh,   Nv, Kv, Nvp, Kvp, 1);
  cvt(Wv2,  Wv2h,  Nv2, Kv2, Nv2p, Kv2p, 1);
  cvt(Wfc1, Wfc1h, Nf1, Kf1, Nf1p, Kf1p, 10);
  cvt(Wfc2, Wfc2h, Nf2, Kf2, Nf2p, Kf2p, 10);
  cvt(x2, x2h, B, Q * M, B, K2p, 1);                   // rows = batch
  { dim3 g((K1p + 255) / 256, B, 1);
    k_gather_e1h<<<g, b256, 0, stream>>>(x1, emb, e1h, K1p); }

  // --- w = e1 @ Ww^T + bw (the 102-GFLOP GEMM) ; u = x2 @ Wu^T + bu ---
  { dim3 g(N1p / 64, B / 128, 1);
    k_gemm_f16<false, false><<<g, b128, 0, stream>>>(
        e1h, 0, K1p, Wwh, 0, K1p, bw, 0, N1, w_out, 0, N1p, N1p, K1p); }
  { dim3 g(N2p / 64, B / 128, 1);
    k_gemm_f16<false, false><<<g, b128, 0, stream>>>(
        x2h, 0, K2p, Wuh, 0, K2p, bu, 0, N2, u_out, 0, N2p, N2p, K2p); }

  // --- s = mean tanh(w+u) ; v = s @ Wv^T + bv ; alpha softmax ---
  k_coattn1<<<dim3(B), b512, 0, stream>>>(w_out, N1p, u_out, N2p, sh, Kvp);
  { dim3 g(Nvp / 64, B / 128, 1);
    k_gemm_f16<false, false><<<g, b128, 0, stream>>>(
        sh, 0, Kvp, Wvh, 0, Kvp, bv, 0, Nv, v_out, 0, Nvp, Nvp, Kvp); }
  k_alpha<<<dim3((B * P + 255) / 256), b256, 0, stream>>>(v_out, x1len, x2len, alpha, B * P);

  // --- inp = einsum(bpq,bpm->bqm), masked ---
  k_inp<<<dim3(B), b256, 0, stream>>>(x1, emb, alpha, x2len, inp, inph);

  // --- w2 = inp @ Ww2^T + bw2 ; u2 = x2 @ Wu2^T + bu2 ---
  { dim3 g(N2p / 64, B / 128, 1);
    k_gemm_f16<false, false><<<g, b128, 0, stream>>>(
        inph, 0, K2p, Ww2h, 0, K2p, bw2, 0, N2, w2o, 0, N2p, N2p, K2p);
    k_gemm_f16<false, false><<<g, b128, 0, stream>>>(
        x2h, 0, K2p, Wu2h, 0, K2p, bu2, 0, N2, u2o, 0, N2p, N2p, K2p); }

  // --- s2, broadcast, v2 ---
  k_s2<<<dim3((B * Q + 255) / 256), b256, 0, stream>>>(w2o, u2o, s2, B * Q);
  k_s2h<<<dim3(B), dim3(Kv2p), 0, stream>>>(s2, s2h);
  { dim3 g(Nv2p / 64, B / 128, 1);
    k_gemm_f16<false, false><<<g, b128, 0, stream>>>(
        s2h, 0, Kv2p, Wv2h, 0, Kv2p, bv2, 0, Nv2, v2o, 0, Nv2p, Nv2p, Kv2p); }

  // --- alpha2, shared, x_mlp, xg ---
  k_stage2<<<dim3(B), b128, 0, stream>>>(v2o, x2len, inp, x2, xgh);

  // --- 10 MLP heads: h1 = relu(xg @ Wfc1^T + bfc1)  -> f16, grouped grid.z ---
  { dim3 g(Nf1p / 64, B / 128, 10);
    k_gemm_f16<true, true><<<g, b128, 0, stream>>>(
        xgh, Kf1p, 10 * Kf1p, Wfc1h, (long long)Nf1p * Kf1p, Kf1p,
        bfc1, Nf1, Nf1, h1h, Nf1p, 10 * Nf1p, Nf1p, Kf1p); }

  // --- h2 = relu(h1 @ Wfc2^T + bfc2) written straight into d_out [B,10,800] ---
  { dim3 g(Nf2p / 64, B / 128, 10);
    k_gemm_f16<true, false><<<g, b128, 0, stream>>>(
        h1h, Nf1p, 10 * Nf1p, Wfc2h, (long long)Nf2p * Kf2p, Kf2p,
        bfc2, Nf2, Nf2, out, Nf2, 10 * Nf2, Nf2, Kf2p); }

  // --- final softmax over 800 ---
  k_softmax800<<<dim3(B * 10), b256, 0, stream>>>(out);
}
